// PatternBranch_31121333027530
// MI455X (gfx1250) — matrix operand order repaired and checked by reading, not compiled, not run on hardware
//
#include <hip/hip_runtime.h>
#include <hip/hip_bf16.h>

// ---------------------------------------------------------------------------
// PatternBranch for MI455X (gfx1250, wave32, WMMA).
// bf16 activations/weights + f32-accumulate WMMA GEMMs (traffic-bound:
// bf16 halves the 268MB feats0 stream). Split-K with f32 atomics fills the
// WGPs. GEMM tiles are staged memory->LDS with GLOBAL_LOAD_ASYNC_TO_LDS_B128
// (ASYNCcnt-tracked, double-buffered) directly into WMMA operand layout.
// ---------------------------------------------------------------------------

typedef __bf16  bf16_t;
typedef bf16_t  bf16x16 __attribute__((ext_vector_type(16)));
typedef float   f32x8   __attribute__((ext_vector_type(8)));

// Problem dims (fixed by reference)
#define BATCH 1024
#define HH    32
#define WW    32
#define CIN   3
#define CC    128
#define KPAT  16
#define HID   256
#define NCLS  3
#define KBASE (HH * WW * CC)     // 131072
#define KSEL  (HH * WW * KPAT)   // 16384

// ---------------------------------------------------------------------------
// 1) f32 [K][256] -> bf16 [256][K] transpose-convert (coalesced via LDS tile)
// ---------------------------------------------------------------------------
__global__ __launch_bounds__(256) void transpose_convert_kernel(
    const float* __restrict__ W, bf16_t* __restrict__ Wt, int K) {
  __shared__ float tile[32][33];
  int k0 = blockIdx.x * 32, n0 = blockIdx.y * 32;
  int tx = threadIdx.x, ty = threadIdx.y;   // block (32, 8)
#pragma unroll
  for (int i = 0; i < 32; i += 8)
    tile[ty + i][tx] = W[(size_t)(k0 + ty + i) * HID + (n0 + tx)];
  __syncthreads();
#pragma unroll
  for (int i = 0; i < 32; i += 8)
    Wt[(size_t)(n0 + ty + i) * K + (k0 + tx)] = (bf16_t)tile[tx][ty + i];
}

// ---------------------------------------------------------------------------
// 2) Conv3x3 SAME + bias + ReLU -> feats (bf16, K-flattened (h*32+w)*128+c),
//    fused matcher partial: atomicAdd per block of sum_{w,c} relu*w_match[c].
// ---------------------------------------------------------------------------
__global__ __launch_bounds__(256) void conv_relu_kernel(
    const float* __restrict__ in, const float* __restrict__ Wfe,
    const float* __restrict__ bfe, const float* __restrict__ wm,
    bf16_t* __restrict__ feats, float* __restrict__ match_acc) {
  __shared__ float sW[3 * 3 * CIN * CC];
  __shared__ float sIn[3][34][CIN];
  __shared__ float sM[CC];
  __shared__ float sRed[256];

  int tid = threadIdx.x;
  int b = blockIdx.x >> 5;
  int h = blockIdx.x & 31;

  for (int t = tid; t < 3 * 3 * CIN * CC; t += 256) sW[t] = Wfe[t];
  if (tid < CC) sM[tid] = wm[tid];
  for (int t = tid; t < 3 * 34 * CIN; t += 256) {
    int ky = t / (34 * CIN), rem = t % (34 * CIN);
    int wp = rem / CIN, ci = rem % CIN;
    int gh = h - 1 + ky, gw = wp - 1;
    float v = 0.f;
    if (gh >= 0 && gh < HH && gw >= 0 && gw < WW)
      v = in[(((size_t)b * HH + gh) * WW + gw) * CIN + ci];
    sIn[ky][wp][ci] = v;
  }
  __syncthreads();

  int c = tid & (CC - 1);
  int wsel = tid >> 7;
  float bias = bfe[c];
  float wmc = sM[c];
  float msum = 0.f;

  for (int w = wsel; w < WW; w += 2) {
    float acc = bias;
#pragma unroll
    for (int ky = 0; ky < 3; ++ky)
#pragma unroll
      for (int kx = 0; kx < 3; ++kx)
#pragma unroll
        for (int ci = 0; ci < CIN; ++ci)
          acc = fmaf(sIn[ky][w + kx][ci], sW[((ky * 3 + kx) * CIN + ci) * CC + c], acc);
    float rv = acc > 0.f ? acc : 0.f;
    feats[((size_t)b * (HH * WW) + h * WW + w) * CC + c] = (bf16_t)rv;
    msum = fmaf(rv, wmc, msum);
  }

  sRed[tid] = msum;
  __syncthreads();
  for (int s = 128; s > 0; s >>= 1) {
    if (tid < s) sRed[tid] += sRed[tid + s];
    __syncthreads();
  }
  if (tid == 0) atomicAdd(match_acc + b, sRed[0]);
}

// ---------------------------------------------------------------------------
// 3) Gather pat channels: sel[b][pos*16+j] = feats[b][pos*128 + pat_idx[j]]
// ---------------------------------------------------------------------------
__global__ __launch_bounds__(256) void gather_sel_kernel(
    const bf16_t* __restrict__ feats, const int* __restrict__ idx,
    bf16_t* __restrict__ sel) {
  int g = blockIdx.x * 256 + threadIdx.x;
  int b = g >> 14;
  int rem = g & (KSEL - 1);
  int pos = rem >> 4, j = rem & 15;
  sel[(size_t)b * KSEL + rem] = feats[(size_t)b * KBASE + pos * CC + idx[j]];
}

// ---------------------------------------------------------------------------
// 4) bf16 WMMA GEMM, split-K:  out[M=1024][256] += A[M][K] * Wt[N][K]^T
//    Block tile 128(M) x 64(N), K staged 32-wide, DOUBLE-BUFFERED via
//    GLOBAL_LOAD_ASYNC_TO_LDS_B128 (ASYNCcnt). LDS holds operands already in
//    the ISA WMMA lane layout so each wave operand read is one 32B ds load:
//      A (16-bit 16x32): lane<16  -> M=lane,    halves = K[0..7],K[16..23]
//                        lane>=16 -> M=lane-16, halves = K[8..15],K[24..31]
//      B (16-bit 32x16): lane<16  -> N=lane,    halves = K[0..7],K[16..23]
//                        lane>=16 -> N=lane-16, halves = K[8..15],K[24..31]
//    Tile = A 8KB (512 x 16B chunks) + B 4KB (256 x 16B chunks) = 768 chunks
//    = exactly 3 async B128 ops per thread -> uniform s_wait_asynccnt 3.
// ---------------------------------------------------------------------------
#define A_ELEMS (8 * 32 * 16)               // 4096 bf16 = 8 KB
#define B_ELEMS (4 * 32 * 16)               // 2048 bf16 = 4 KB
#define TILE_ELEMS (A_ELEMS + B_ELEMS)      // 12 KB per buffer

__global__ __launch_bounds__(256) void gemm_bf16_wmma_kernel(
    const bf16_t* __restrict__ A, const bf16_t* __restrict__ Wt,
    float* __restrict__ out, int K, int kChunk) {
  __shared__ bf16_t sT[2][TILE_ELEMS];

  int tid  = threadIdx.x;
  int m0   = blockIdx.y * 128;
  int n0   = blockIdx.x * 64;
  int wave = tid >> 5, lane = tid & 31;

  // Precompute this thread's 3 staging chunks (global base, LDS dest offset).
  unsigned long long gbase[3];
  unsigned ldsoff[3];
#pragma unroll
  for (int i = 0; i < 3; ++i) {
    int cid = i * 256 + tid;                  // 0..767
    const bf16_t* src;
    unsigned dst;                             // element offset within tile
    if (cid < 512) {                          // A chunk: row r, quarter q
      int r = cid >> 2, q = cid & 3;          // q: K[8q .. 8q+7]
      int msub = r >> 4, rl = r & 15;
      int lane_add = (q & 1) << 4;            // K[8..15],K[24..31] -> lane+16
      int slot     = (q >> 1) << 3;           // K[16..31] -> halves 8..15
      src = A + (size_t)(m0 + r) * K + q * 8;
      dst = (unsigned)(((msub * 32) + rl + lane_add) * 16 + slot);
    } else {                                  // B chunk
      int cb = cid - 512;
      int n = cb >> 2, q = cb & 3;
      int nsub = n >> 4, nl = n & 15;
      int lane_add = (q & 1) << 4;            // K[8..15],K[24..31] -> lane+16
      int slot     = (q >> 1) << 3;           // K[16..31] -> halves 8..15
      src = Wt + (size_t)(n0 + n) * K + q * 8;
      dst = (unsigned)(A_ELEMS + ((nsub * 32) + nl + lane_add) * 16 + slot);
    }
    gbase[i]  = (unsigned long long)src;
    // generic LDS pointer: low 32 bits are the ds-usable LDS byte address
    ldsoff[i] = (unsigned)(size_t)&sT[0][dst];
  }

  f32x8 acc[4];
#pragma unroll
  for (int ns = 0; ns < 4; ++ns)
    acc[ns] = (f32x8){0.f, 0.f, 0.f, 0.f, 0.f, 0.f, 0.f, 0.f};

  int kb0    = blockIdx.z * kChunk;
  int nIters = kChunk >> 5;

  // Prologue: async-stage tile 0 into buffer 0 (3 ops -> ASYNCcnt = 3)
#pragma unroll
  for (int i = 0; i < 3; ++i) {
    unsigned long long g = gbase[i] + (unsigned long long)kb0 * 2u;
    unsigned l = ldsoff[i];
    asm volatile("global_load_async_to_lds_b128 %0, %1, off"
                 :: "v"(l), "v"(g) : "memory");
  }

  for (int it = 0; it < nIters; ++it) {
    int buf = it & 1;
    if (it + 1 < nIters) {
      int kb = kb0 + ((it + 1) << 5);
#pragma unroll
      for (int i = 0; i < 3; ++i) {
        unsigned long long g = gbase[i] + (unsigned long long)kb * 2u;
        unsigned l = ldsoff[i] + (unsigned)((buf ^ 1) * (TILE_ELEMS * 2));
        asm volatile("global_load_async_to_lds_b128 %0, %1, off"
                     :: "v"(l), "v"(g) : "memory");
      }
      // in-order completion: <=3 outstanding == current tile fully in LDS
      asm volatile("s_wait_asynccnt 0x3" ::: "memory");
    } else {
      asm volatile("s_wait_asynccnt 0x0" ::: "memory");
    }
    __syncthreads();   // all waves' tile-it writes visible

    const bf16_t* base = &sT[buf][0];
    bf16x16 av = *(const bf16x16*)(base + (wave * 32 + lane) * 16);
#pragma unroll
    for (int ns = 0; ns < 4; ++ns) {
      bf16x16 bv = *(const bf16x16*)(base + A_ELEMS + (ns * 32 + lane) * 16);
      acc[ns] = __builtin_amdgcn_wmma_f32_16x16x32_bf16(
          false, av, false, bv, (short)0, acc[ns], false, false);
    }
    __syncthreads();   // all reads done before this buffer is overwritten
  }

  // C/D layout: VGPR r, lane L: M = r + (L<16 ? 0 : 8), N = L%16
#pragma unroll
  for (int ns = 0; ns < 4; ++ns)
#pragma unroll
    for (int rr = 0; rr < 8; ++rr) {
      int row = m0 + wave * 16 + rr + ((lane >> 4) << 3);
      int col = n0 + ns * 16 + (lane & 15);
      atomicAdd(out + (size_t)row * HID + col, acc[ns][rr]);
    }
}

// ---------------------------------------------------------------------------
// 5) Head: routing select + bias + ReLU + [256 -> 3] layer 2 + merge.
// ---------------------------------------------------------------------------
__global__ __launch_bounds__(256) void head_kernel(
    const float* __restrict__ hidB, const float* __restrict__ hidP,
    const float* __restrict__ match_acc, const float* __restrict__ bmatch,
    const float* __restrict__ bB1, const float* __restrict__ WB2,
    const float* __restrict__ bB2, const float* __restrict__ bP1,
    const float* __restrict__ WP2, const float* __restrict__ bP2,
    float* __restrict__ out) {
  __shared__ float sR[3 * 256];
  int b = blockIdx.x, tid = threadIdx.x;
  bool m = (match_acc[b] * (1.f / 1024.f) + bmatch[0]) > 0.f;
  const float* hid = m ? hidP : hidB;
  const float* b1  = m ? bP1 : bB1;
  const float* W2  = m ? WP2 : WB2;
  const float* b2  = m ? bP2 : bB2;

  float hv = hid[(size_t)b * HID + tid] + b1[tid];
  hv = hv > 0.f ? hv : 0.f;
  sR[tid]       = hv * W2[tid * NCLS + 0];
  sR[256 + tid] = hv * W2[tid * NCLS + 1];
  sR[512 + tid] = hv * W2[tid * NCLS + 2];
  __syncthreads();
  for (int s = 128; s > 0; s >>= 1) {
    if (tid < s) {
      sR[tid]       += sR[tid + s];
      sR[256 + tid] += sR[256 + tid + s];
      sR[512 + tid] += sR[512 + tid + s];
    }
    __syncthreads();
  }
  if (tid < NCLS) out[b * NCLS + tid] = sR[tid * 256] + b2[tid];
}

// ---------------------------------------------------------------------------
// Launch
// ---------------------------------------------------------------------------
extern "C" void kernel_launch(void* const* d_in, const int* in_sizes, int n_in,
                              void* d_out, int out_size, void* d_ws, size_t ws_size,
                              hipStream_t stream) {
  const float* inputs   = (const float*)d_in[0];
  const float* W_fe     = (const float*)d_in[1];
  const float* b_fe     = (const float*)d_in[2];
  const float* w_match  = (const float*)d_in[3];
  const float* b_match  = (const float*)d_in[4];
  const int*   pat_idx  = (const int*)d_in[5];
  const float* W_pat1   = (const float*)d_in[6];
  const float* b_pat1   = (const float*)d_in[7];
  const float* W_pat2   = (const float*)d_in[8];
  const float* b_pat2   = (const float*)d_in[9];
  const float* W_base1  = (const float*)d_in[10];
  const float* b_base1  = (const float*)d_in[11];
  const float* W_base2  = (const float*)d_in[12];
  const float* b_base2  = (const float*)d_in[13];
  float* out = (float*)d_out;

  // Workspace carve-up (~380 MB total)
  char* ws = (char*)d_ws;
  size_t off = 0;
  bf16_t* feats = (bf16_t*)(ws + off); off += (size_t)BATCH * KBASE * 2;   // 268 MB
  bf16_t* sel   = (bf16_t*)(ws + off); off += (size_t)BATCH * KSEL  * 2;   // 33.5 MB
  bf16_t* Wb1t  = (bf16_t*)(ws + off); off += (size_t)KBASE * HID   * 2;   // 67 MB
  bf16_t* Wp1t  = (bf16_t*)(ws + off); off += (size_t)KSEL  * HID   * 2;   // 8.4 MB
  float*  hidB  = (float*)(ws + off);  off += (size_t)BATCH * HID   * 4;   // 1 MB
  float*  hidP  = (float*)(ws + off);  off += (size_t)BATCH * HID   * 4;   // 1 MB
  float*  macc  = (float*)(ws + off);  off += (size_t)BATCH * 4;

  // Zero split-K accumulators + matcher accumulator (contiguous region)
  hipMemsetAsync(hidB, 0, (size_t)BATCH * HID * 4 * 2 + (size_t)BATCH * 4, stream);

  // Weight transpose-converts (independent)
  transpose_convert_kernel<<<dim3(KBASE / 32, HID / 32), dim3(32, 8), 0, stream>>>(
      W_base1, Wb1t, KBASE);
  transpose_convert_kernel<<<dim3(KSEL / 32, HID / 32), dim3(32, 8), 0, stream>>>(
      W_pat1, Wp1t, KSEL);

  // Conv + ReLU + fused matcher partial sums
  conv_relu_kernel<<<BATCH * HH, 256, 0, stream>>>(
      inputs, W_fe, b_fe, w_match, feats, macc);

  // Pat-channel gather
  gather_sel_kernel<<<(BATCH * KSEL) / 256, 256, 0, stream>>>(feats, pat_idx, sel);

  // GEMM1 (base): M=1024, N=256, K=131072, split-K 32 -> 1024 blocks
  gemm_bf16_wmma_kernel<<<dim3(HID / 64, BATCH / 128, 32), 256, 0, stream>>>(
      feats, Wb1t, hidB, KBASE, KBASE / 32);
  // GEMM1 (pat): K=16384, split-K 8 -> 256 blocks
  gemm_bf16_wmma_kernel<<<dim3(HID / 64, BATCH / 128, 8), 256, 0, stream>>>(
      sel, Wp1t, hidP, KSEL, KSEL / 8);

  // Layer 2 + routed merge
  head_kernel<<<BATCH, 256, 0, stream>>>(
      hidB, hidP, macc, b_match, b_base1, W_base2, b_base2,
      b_pat1, W_pat2, b_pat2, out);
}
